// VectorQuantizerEMA_25872882991419
// MI455X (gfx1250) — matrix operand order repaired and checked by reading, not compiled, not run on hardware
//
#include <hip/hip_runtime.h>
#include <hip/hip_bf16.h>

// ---------------- problem constants ----------------
#define VQ_N 32768
#define VQ_K 8192
#define VQ_D 64
#define VQ_DECAY 0.99f
#define VQ_EPS 1e-5f

// ---------------- WMMA types (CDNA5 / gfx1250, wave32) ----------------
typedef __attribute__((ext_vector_type(16))) __bf16 v16bf;
typedef __attribute__((ext_vector_type(8)))  float  v8f;

union V16U {
    v16bf v;
    uint4 q[2];
};

__device__ __forceinline__ unsigned short f2bf(float f) {
    // round-to-nearest-even f32 -> bf16
    unsigned int u = __float_as_uint(f);
    unsigned int r = u + 0x7FFFu + ((u >> 16) & 1u);
    return (unsigned short)(r >> 16);
}

// ---------------- kernel 1a: X -> bf16 ----------------
__global__ __launch_bounds__(256) void vq_convert_x(const float* __restrict__ X,
                                                    unsigned short* __restrict__ Xbf,
                                                    int n) {
    int i = blockIdx.x * 256 + threadIdx.x;
    if (i < n) Xbf[i] = f2bf(X[i]);
}

// ---------------- kernel 1b: E -> bf16 + ||e||^2 ----------------
// 8 waves per block, one wave per codebook row (D=64 -> 2 elems/lane).
__global__ __launch_bounds__(256) void vq_prep_e(const float* __restrict__ E,
                                                 unsigned short* __restrict__ Ebf,
                                                 float* __restrict__ eNorm) {
    const int wave = threadIdx.x >> 5;
    const int lane = threadIdx.x & 31;
    const int k = blockIdx.x * 8 + wave;
    const float v0 = E[(size_t)k * VQ_D + lane];
    const float v1 = E[(size_t)k * VQ_D + 32 + lane];
    Ebf[(size_t)k * VQ_D + lane]      = f2bf(v0);
    Ebf[(size_t)k * VQ_D + 32 + lane] = f2bf(v1);
    float s = v0 * v0 + v1 * v1;
    #pragma unroll
    for (int off = 16; off >= 1; off >>= 1)
        s += __shfl_xor(s, off, 32);
    if (lane == 0) eNorm[k] = s;
}

// ---------------- kernel 2: WMMA distance + argmin ----------------
// Block: 256 threads = 8 waves, 128 rows (16 rows per wave).
// E staged through LDS in 128-code (16 KB) chunks via DOUBLE-BUFFERED
// gfx1250 async-to-LDS copies (ASYNCcnt path); A held in registers.
#define CHUNK_CODES 128

// Issue one chunk's async copy: 16 KB of bf16 codes (4 x b128 per thread)
// plus 512 B of ||e||^2 (1 x b32 for waves 0-3). Tracked on ASYNCcnt.
__device__ __forceinline__ void vq_async_copy_chunk(const unsigned short* __restrict__ Ebf,
                                                    const float* __restrict__ eNorm,
                                                    unsigned short* ldsEbuf,
                                                    float* ldsNbuf,
                                                    int chunk, int tid) {
    const char* gsrc = (const char*)(Ebf + (size_t)chunk * CHUNK_CODES * VQ_D);
    #pragma unroll
    for (int i = 0; i < 4; ++i) {
        unsigned lds_dst =
            (unsigned)(uintptr_t)((char*)ldsEbuf + tid * 16 + i * 4096);
        unsigned long long ga =
            (unsigned long long)(uintptr_t)(gsrc + tid * 16 + i * 4096);
        asm volatile("global_load_async_to_lds_b128 %0, %1, off"
                     :: "v"(lds_dst), "v"(ga) : "memory");
    }
    if (tid < 128) {  // wave-uniform (waves 0-3)
        unsigned lds_dst = (unsigned)(uintptr_t)(ldsNbuf + tid);
        unsigned long long ga =
            (unsigned long long)(uintptr_t)(eNorm + chunk * CHUNK_CODES + tid);
        asm volatile("global_load_async_to_lds_b32 %0, %1, off"
                     :: "v"(lds_dst), "v"(ga) : "memory");
    }
}

__device__ __forceinline__ void vq_wait_async() {
    asm volatile("s_wait_asynccnt 0x0" ::: "memory");
}

__global__ __launch_bounds__(256) void vq_argmin_wmma(const unsigned short* __restrict__ Xbf,
                                                      const unsigned short* __restrict__ Ebf,
                                                      const float* __restrict__ eNorm,
                                                      int* __restrict__ indices) {
    __shared__ unsigned short ldsE[2][CHUNK_CODES * VQ_D];  // 2 x 16 KB
    __shared__ float          ldsN[2][CHUNK_CODES];         // 2 x 512 B

    const int tid  = threadIdx.x;
    const int wave = tid >> 5;
    const int lane = tid & 31;
    const int l    = lane & 15;   // column / row-within-tile index
    const int h    = lane >> 4;   // half of the wave (K-slice selector)
    const int mBase = blockIdx.x * 128 + wave * 16;
    const int row   = mBase + l;  // A row supplied by this lane (both halves)

    // ---- load A fragments (16x64 bf16) in CDNA5 A-layout, kept in regs ----
    // lane l   : K = 0..7   (VGPR0-3), K = 16..23 (VGPR4-7)
    // lane 16+l: K = 8..15  (VGPR0-3), K = 24..31 (VGPR4-7)
    const uint4* xrow = reinterpret_cast<const uint4*>(Xbf + (size_t)row * VQ_D);
    V16U a0, a1;
    a0.q[0] = xrow[0 + h];  // elems  8h ..  8h+7   (d0 = 0)
    a0.q[1] = xrow[2 + h];  // elems 16+8h .. +7
    a1.q[0] = xrow[4 + h];  // elems 32+8h .. +7    (d0 = 32)
    a1.q[1] = xrow[6 + h];  // elems 48+8h .. +7

    float minv[8];
    int   mini[8];
    #pragma unroll
    for (int v = 0; v < 8; ++v) { minv[v] = 3.4e38f; mini[v] = 0; }

    const int numChunks = VQ_K / CHUNK_CODES;  // 64

    // prologue: async-stage chunk 0 into buffer 0
    vq_async_copy_chunk(Ebf, eNorm, ldsE[0], ldsN[0], 0, tid);
    vq_wait_async();
    __syncthreads();

    for (int c = 0; c < numChunks; ++c) {
        const unsigned short* eb = ldsE[c & 1];
        const float*          nb = ldsN[c & 1];

        // overlap: kick off next chunk's async copy into the other buffer
        // (safe: that buffer was last read in iteration c-1, barrier passed)
        if (c + 1 < numChunks)
            vq_async_copy_chunk(Ebf, eNorm, ldsE[(c + 1) & 1], ldsN[(c + 1) & 1],
                                c + 1, tid);

        #pragma unroll
        for (int t = 0; t < CHUNK_CODES / 16; ++t) {
            const int cl = t * 16 + l;  // local code = this lane's score column
            // ---- B fragment (32x16 bf16) from LDS: same K order as A ----
            // lane l: K(d) = d0+0..7, d0+16..23 ; lane 16+l: K(d) = d0+8..15, d0+24..31 ; column N = l
            const uint4* erow = reinterpret_cast<const uint4*>(eb + cl * VQ_D);
            V16U b0, b1;
            b0.q[0] = erow[0 + h];
            b0.q[1] = erow[2 + h];
            b1.q[0] = erow[4 + h];
            b1.q[1] = erow[6 + h];

            v8f acc = {};
            acc = __builtin_amdgcn_wmma_f32_16x16x32_bf16(false, a0.v, false, b0.v,
                                                          (short)0, acc, false, false);
            acc = __builtin_amdgcn_wmma_f32_16x16x32_bf16(false, a1.v, false, b1.v,
                                                          (short)0, acc, false, false);

            const float en = nb[cl];
            const int   gc = c * CHUNK_CODES + cl;
            #pragma unroll
            for (int v = 0; v < 8; ++v) {
                // dist = ||e||^2 - 2 x.e   (||x||^2 is row-constant: argmin-invariant)
                const float dist = en - 2.0f * acc[v];
                if (dist < minv[v]) { minv[v] = dist; mini[v] = gc; }
            }
        }

        // chunk c+1 landed (own lanes) + everyone done reading chunk c
        if (c + 1 < numChunks) vq_wait_async();
        __syncthreads();
    }

    // ---- per-row argmin across the 16 lanes of each half (tie -> lower idx) ----
    #pragma unroll
    for (int v = 0; v < 8; ++v) {
        float mv = minv[v];
        int   mi = mini[v];
        #pragma unroll
        for (int off = 8; off >= 1; off >>= 1) {
            const float ov = __shfl_xor(mv, off, 32);
            const int   oi = __shfl_xor(mi, off, 32);
            if (ov < mv || (ov == mv && oi < mi)) { mv = ov; mi = oi; }
        }
        if (l == 0) indices[mBase + v + 8 * h] = mi;  // lanes<16: M=v ; lanes>=16: M=v+8
    }
}

// ---------------- kernel 3: one-hot + counts + dw scatter ----------------
__global__ __launch_bounds__(64) void vq_scatter(const float* __restrict__ X,
                                                 const int* __restrict__ indices,
                                                 float* __restrict__ enc,
                                                 float* __restrict__ counts,
                                                 float* __restrict__ dw) {
    const int i = blockIdx.x;
    const int t = threadIdx.x;
    const int k = indices[i];
    atomicAdd(&dw[(size_t)k * VQ_D + t], X[(size_t)i * VQ_D + t]);
    if (t == 0) {
        enc[(size_t)i * VQ_K + k] = 1.0f;
        atomicAdd(&counts[k], 1.0f);
    }
}

// ---------------- kernel 4: cs_pre + global sum n ----------------
__global__ __launch_bounds__(256) void vq_cs_pre(const float* __restrict__ ecs,
                                                 const float* __restrict__ counts,
                                                 float* __restrict__ cs_pre,
                                                 float* __restrict__ nsum) {
    __shared__ float red[256];
    const int k = blockIdx.x * 256 + threadIdx.x;
    const float cs = ecs[k] * VQ_DECAY + (1.0f - VQ_DECAY) * counts[k];
    cs_pre[k] = cs;
    red[threadIdx.x] = cs;
    __syncthreads();
    for (int s = 128; s > 0; s >>= 1) {
        if (threadIdx.x < s) red[threadIdx.x] += red[threadIdx.x + s];
        __syncthreads();
    }
    if (threadIdx.x == 0) atomicAdd(nsum, red[0]);
}

// ---------------- kernel 5: finalize outputs ----------------
__global__ __launch_bounds__(64) void vq_finalize(const float* __restrict__ E,
                                                  const float* __restrict__ emaW,
                                                  const float* __restrict__ dw,
                                                  const float* __restrict__ cs_pre,
                                                  const float* __restrict__ nsum,
                                                  float* __restrict__ out_cb,
                                                  float* __restrict__ out_cs,
                                                  float* __restrict__ out_emaw,
                                                  float* __restrict__ out_emb) {
    const int k = blockIdx.x;
    const int t = threadIdx.x;
    const float n  = *nsum;
    const float cs = (cs_pre[k] + VQ_EPS) / (n + (float)VQ_K * VQ_EPS) * n;
    const size_t o = (size_t)k * VQ_D + t;
    const float e = emaW[o] * VQ_DECAY + (1.0f - VQ_DECAY) * dw[o];
    out_cb[o]   = E[o];
    out_emaw[o] = e;
    out_emb[o]  = e / cs;
    if (t == 0) out_cs[k] = cs;
}

// ---------------- host-side launcher ----------------
static inline size_t alignUp(size_t x, size_t a) { return (x + a - 1) & ~(a - 1); }

extern "C" void kernel_launch(void* const* d_in, const int* in_sizes, int n_in,
                              void* d_out, int out_size, void* d_ws, size_t ws_size,
                              hipStream_t stream) {
    (void)in_sizes; (void)n_in; (void)out_size; (void)ws_size;

    const float* X    = (const float*)d_in[0];  // [N, D]
    const float* E    = (const float*)d_in[1];  // [K, D]
    const float* emaW = (const float*)d_in[2];  // [K, D]
    const float* ecs  = (const float*)d_in[3];  // [K]

    // output layout: encodings | codebook | cluster_size | ema_w_new | embedding_new
    float* out      = (float*)d_out;
    float* out_enc  = out;
    float* out_cb   = out_enc + (size_t)VQ_N * VQ_K;
    float* out_cs   = out_cb + (size_t)VQ_K * VQ_D;
    float* out_emaw = out_cs + VQ_K;
    float* out_emb  = out_emaw + (size_t)VQ_K * VQ_D;

    // workspace layout
    char* ws = (char*)d_ws;
    unsigned short* Xbf = (unsigned short*)ws; ws += alignUp((size_t)VQ_N * VQ_D * 2, 256);
    unsigned short* Ebf = (unsigned short*)ws; ws += alignUp((size_t)VQ_K * VQ_D * 2, 256);
    float* eNorm  = (float*)ws; ws += alignUp((size_t)VQ_K * 4, 256);
    int*   idx    = (int*)ws;   ws += alignUp((size_t)VQ_N * 4, 256);
    float* counts = (float*)ws; ws += alignUp((size_t)VQ_K * 4, 256);
    float* dw     = (float*)ws; ws += alignUp((size_t)VQ_K * VQ_D * 4, 256);
    float* cs_pre = (float*)ws; ws += alignUp((size_t)VQ_K * 4, 256);
    float* nsum   = (float*)ws; ws += 256;

    // per-call zero init (graph-capturable)
    hipMemsetAsync(out_enc, 0, (size_t)VQ_N * VQ_K * sizeof(float), stream);
    hipMemsetAsync(counts, 0, (size_t)VQ_K * sizeof(float), stream);
    hipMemsetAsync(dw, 0, (size_t)VQ_K * VQ_D * sizeof(float), stream);
    hipMemsetAsync(nsum, 0, sizeof(float), stream);

    vq_convert_x<<<(VQ_N * VQ_D + 255) / 256, 256, 0, stream>>>(X, Xbf, VQ_N * VQ_D);
    vq_prep_e<<<VQ_K / 8, 256, 0, stream>>>(E, Ebf, eNorm);
    vq_argmin_wmma<<<VQ_N / 128, 256, 0, stream>>>(Xbf, Ebf, eNorm, idx);
    vq_scatter<<<VQ_N, 64, 0, stream>>>(X, idx, out_enc, counts, dw);
    vq_cs_pre<<<VQ_K / 256, 256, 0, stream>>>(ecs, counts, cs_pre, nsum);
    vq_finalize<<<VQ_K, 64, 0, stream>>>(E, emaW, dw, cs_pre, nsum,
                                         out_cb, out_cs, out_emaw, out_emb);
}
